// Decoder_23562190586172
// MI455X (gfx1250) — matrix operand order repaired and checked by reading, not compiled, not run on hardware
//
#include <hip/hip_runtime.h>
#include <hip/hip_bf16.h>
#include <math.h>

typedef __bf16 bf16_t;
typedef __attribute__((ext_vector_type(16))) __bf16 v16bf;
typedef __attribute__((ext_vector_type(8)))  __bf16 v8bf;
typedef __attribute__((ext_vector_type(8)))  float  v8f;

#define BATCH 16
#define LSEQ  400
#define TSTEPS 200
#define NMELS 80
#define ENC   512
#define PRE   256
#define QHD   1024
#define DHD   1024
#define ATTND 128
#define NFILT 32
#define KCONV 31
#define GATES 4096   // 4*1024
#define KQ    1792   // 256 + 512 + 1024
#define KD    2560   // 512 + 1024 + 1024

__device__ __forceinline__ float sigf(float x) { return 1.0f / (1.0f + __expf(-x)); }

// ---------------- WMMA fragment loaders (bf16, 16x16x32) ----------------
// A (16 x K row-major): lane<16 -> row M=lane, K chunks [k0..k0+7],[k0+16..k0+23]
//                       lane>=16 -> row M=lane-16, chunks shifted +8
__device__ __forceinline__ v16bf load_a_frag(const bf16_t* A, int strideA, int k0) {
  int lane = threadIdx.x & 31;
  int row  = lane & 15;
  int kb   = k0 + ((lane >> 4) << 3);
  const bf16_t* p = A + (size_t)row * strideA + kb;
  v8bf lo = *(const v8bf*)p;
  v8bf hi = *(const v8bf*)(p + 16);
  v16bf r;
#pragma unroll
  for (int i = 0; i < 8; ++i) { r[i] = lo[i]; r[8 + i] = hi[i]; }
  return r;
}

// B given as W row-major (N x K) == B^T. lane<16 -> col N=n0+lane, K chunks [k0..k0+7],[k0+16..k0+23]
//                                        lane>=16 -> col N=n0+lane-16, chunks shifted +8
__device__ __forceinline__ v16bf load_b_frag(const bf16_t* W, int strideW, int n0, int k0) {
  int lane = threadIdx.x & 31;
  int n    = n0 + (lane & 15);
  int kb   = k0 + ((lane >> 4) << 3);
  const bf16_t* p = W + (size_t)n * strideW + kb;
  v8bf lo = *(const v8bf*)p;
  v8bf hi = *(const v8bf*)(p + 16);
  v16bf r;
#pragma unroll
  for (int i = 0; i < 8; ++i) { r[i] = lo[i]; r[8 + i] = hi[i]; }
  return r;
}

// -------- WMMA GEMM: Z(16*gridY x N) = [A0|A1|A2](16 x Ktot) * W^T(N x Ktot) --------
// One wave per block, two 16-wide N tiles per wave. blockIdx.y = M tile.
__global__ __launch_bounds__(32)
void wmma_gemm_rowA(const bf16_t* A0, int K0, const bf16_t* A1, int K1,
                    const bf16_t* A2, int K2, const bf16_t* W, int Ktot,
                    float* Z, int N) {
  int lane = threadIdx.x & 31;
  int n0 = blockIdx.x * 32;
  int mt = blockIdx.y;
  const bf16_t* segs[3] = { A0 ? A0 + (size_t)mt * 16 * K0 : A0,
                            A1 ? A1 + (size_t)mt * 16 * K1 : A1,
                            A2 ? A2 + (size_t)mt * 16 * K2 : A2 };
  int ks[3] = { K0, K1, K2 };
  v8f c0 = {}; v8f c1 = {};
  int kofs = 0;
#pragma unroll
  for (int s = 0; s < 3; ++s) {
    const bf16_t* A = segs[s];
    int K = ks[s];
    for (int k = 0; k < K; k += 32) {
      v16bf a  = load_a_frag(A, K, k);
      v16bf b0 = load_b_frag(W, Ktot, n0, kofs + k);
      v16bf b1 = load_b_frag(W, Ktot, n0 + 16, kofs + k);
      c0 = __builtin_amdgcn_wmma_f32_16x16x32_bf16(false, a, false, b0, (short)0, c0, false, false);
      c1 = __builtin_amdgcn_wmma_f32_16x16x32_bf16(false, a, false, b1, (short)0, c1, false, false);
    }
    kofs += K;
  }
  // C/D layout: VGPR r, lane<16 -> M=r,N=lane ; lane>=16 -> M=r+8,N=lane-16
  int mbase = mt * 16 + ((lane >> 4) << 3);
  int ncol  = n0 + (lane & 15);
#pragma unroll
  for (int r = 0; r < 8; ++r) {
    Z[(size_t)(mbase + r) * N + ncol]      = c0[r];
    Z[(size_t)(mbase + r) * N + ncol + 16] = c1[r];
  }
}

// ---------------- LSTM pointwise (gate order i,f,g,o) ----------------
__global__ void lstm_point_kernel(const float* __restrict__ Z, const float* __restrict__ bih,
                                  const float* __restrict__ bhh, float* __restrict__ C,
                                  float* __restrict__ H, bf16_t* __restrict__ Hbf) {
  int tid = blockIdx.x * blockDim.x + threadIdx.x;
  if (tid >= BATCH * QHD) return;
  int m = tid >> 10, j = tid & 1023;
  const float* z = Z + (size_t)m * GATES;
  float zi = z[j]        + bih[j]        + bhh[j];
  float zf = z[j + 1024] + bih[j + 1024] + bhh[j + 1024];
  float zg = z[j + 2048] + bih[j + 2048] + bhh[j + 2048];
  float zo = z[j + 3072] + bih[j + 3072] + bhh[j + 3072];
  float c = sigf(zf) * C[tid] + sigf(zi) * tanhf(zg);
  float h = sigf(zo) * tanhf(c);
  C[tid] = c; H[tid] = h; Hbf[tid] = (bf16_t)h;
}

// ---------------- pq = qh @ attn_wq^T  (16 x 128) ----------------
__global__ void pq_kernel(const float* __restrict__ qh, const float* __restrict__ wq,
                          float* __restrict__ pq) {
  int tid = blockIdx.x * blockDim.x + threadIdx.x;
  if (tid >= BATCH * ATTND) return;
  int b = tid >> 7, a = tid & 127;
  const float* x = qh + (size_t)b * QHD;
  const float* w = wq + (size_t)a * QHD;
  float s = 0.f;
  for (int k = 0; k < QHD; ++k) s += x[k] * w[k];
  pq[tid] = s;
}

// ---------------- location conv -> bf16 filters ----------------
__global__ void conv_loc_kernel(const float* __restrict__ aw, const float* __restrict__ aws,
                                const float* __restrict__ cw, bf16_t* __restrict__ locf_bf) {
  int tid = blockIdx.x * blockDim.x + threadIdx.x;
  if (tid >= BATCH * LSEQ * NFILT) return;
  int f = tid & 31;
  int bl = tid >> 5;
  int l = bl % LSEQ, b = bl / LSEQ;
  const float* c0 = cw + f * 2 * KCONV;
  const float* c1 = c0 + KCONV;
  const float* a0 = aw + b * LSEQ;
  const float* a1 = aws + b * LSEQ;
  float s = 0.f;
#pragma unroll
  for (int k = 0; k < KCONV; ++k) {
    int ll = l + k - 15;
    if (ll >= 0 && ll < LSEQ) s += c0[k] * a0[ll] + c1[k] * a1[ll];
  }
  locf_bf[(size_t)bl * NFILT + f] = (bf16_t)s;
}

// ---------------- energies = tanh(pq + loc + pm) @ v ----------------
__global__ void energies_kernel(const float* __restrict__ pq, const float* __restrict__ locE,
                                const float* __restrict__ pm, const float* __restrict__ v,
                                float* __restrict__ energies) {
  int tid = blockIdx.x * blockDim.x + threadIdx.x;
  if (tid >= BATCH * LSEQ) return;
  int b = tid / LSEQ;
  const float* pqb = pq + (size_t)b * ATTND;
  const float* lo  = locE + (size_t)tid * ATTND;
  const float* pmr = pm + (size_t)tid * ATTND;
  float e = 0.f;
#pragma unroll 4
  for (int a = 0; a < ATTND; ++a) e += v[a] * tanhf(pqb[a] + lo[a] + pmr[a]);
  energies[tid] = e;
}

// ---------------- softmax over L + context, one block per batch row ----------------
__global__ __launch_bounds__(512)
void softmax_ctx_kernel(const float* __restrict__ energies, const float* __restrict__ memory,
                        float* __restrict__ aw, float* __restrict__ aws,
                        float* __restrict__ ctx, bf16_t* __restrict__ ctx_bf,
                        float* __restrict__ out_align, int t) {
  __shared__ float w[LSEQ];
  __shared__ float red[512];
  int b = blockIdx.x, tid = threadIdx.x;
  float e = (tid < LSEQ) ? energies[b * LSEQ + tid] : -1e30f;
  red[tid] = e; __syncthreads();
  for (int s = 256; s > 0; s >>= 1) { if (tid < s) red[tid] = fmaxf(red[tid], red[tid + s]); __syncthreads(); }
  float mx = red[0]; __syncthreads();
  float ex = (tid < LSEQ) ? __expf(e - mx) : 0.f;
  red[tid] = ex; __syncthreads();
  for (int s = 256; s > 0; s >>= 1) { if (tid < s) red[tid] += red[tid + s]; __syncthreads(); }
  float inv = 1.0f / red[0];
  if (tid < LSEQ) {
    float wv = ex * inv;
    w[tid] = wv;
    aw[b * LSEQ + tid] = wv;
    aws[b * LSEQ + tid] += wv;
    out_align[(size_t)b * TSTEPS * LSEQ + (size_t)t * LSEQ + tid] = wv;
  }
  __syncthreads();
  const float* mb = memory + (size_t)b * LSEQ * ENC;
  float acc = 0.f;
  for (int l = 0; l < LSEQ; ++l) acc += w[l] * mb[(size_t)l * ENC + tid];
  ctx[b * ENC + tid] = acc;
  ctx_bf[b * ENC + tid] = (bf16_t)acc;
}

// ---------------- output projection + gate ----------------
__global__ void proj_kernel(const float* __restrict__ dh, const float* __restrict__ ctx,
                            const float* __restrict__ pw, const float* __restrict__ pb,
                            const float* __restrict__ gw, const float* __restrict__ gb,
                            float* __restrict__ out_mel, float* __restrict__ out_stop, int t) {
  int tid = blockIdx.x * blockDim.x + threadIdx.x;
  if (tid >= BATCH * (NMELS + 1)) return;
  int b = tid / (NMELS + 1), o = tid % (NMELS + 1);
  const float* h = dh + (size_t)b * DHD;
  const float* c = ctx + (size_t)b * ENC;
  if (o < NMELS) {
    const float* w = pw + (size_t)o * (DHD + ENC);
    float s = pb[o];
    for (int k = 0; k < DHD; ++k) s += w[k] * h[k];
    for (int k = 0; k < ENC; ++k) s += w[DHD + k] * c[k];
    out_mel[(size_t)b * TSTEPS * NMELS + (size_t)t * NMELS + o] = s;
  } else {
    float s = gb[0];
    for (int k = 0; k < DHD; ++k) s += gw[k] * h[k];
    for (int k = 0; k < ENC; ++k) s += gw[DHD + k] * c[k];
    out_stop[(size_t)b * TSTEPS + t] = sigf(s);
  }
}

// ---------------- one-time prep kernels ----------------
__global__ void pack_w_kernel(const float* __restrict__ wih, int Kih,
                              const float* __restrict__ whh, int Khh,
                              bf16_t* __restrict__ outw, int total) {
  int idx = blockIdx.x * blockDim.x + threadIdx.x;
  if (idx >= total) return;
  int K = Kih + Khh;
  int n = idx / K, k = idx % K;
  float v = (k < Kih) ? wih[(size_t)n * Kih + k] : whh[(size_t)n * Khh + (k - Kih)];
  outw[idx] = (bf16_t)v;
}

__global__ void cvt_bf16_kernel(const float* __restrict__ in, bf16_t* __restrict__ outw, int n) {
  int i = blockIdx.x * blockDim.x + threadIdx.x;
  if (i < n) outw[i] = (bf16_t)in[i];
}

__global__ void prenet1_kernel(const float* __restrict__ teacher, const float* __restrict__ w1,
                               float* __restrict__ h1) {
  int tid = blockIdx.x * blockDim.x + threadIdx.x;
  if (tid >= BATCH * TSTEPS * PRE) return;
  int i = tid / PRE, j = tid % PRE;
  const float* x = teacher + (size_t)i * NMELS;
  const float* w = w1 + (size_t)j * NMELS;
  float s = 0.f;
#pragma unroll
  for (int k = 0; k < NMELS; ++k) s += x[k] * w[k];
  h1[tid] = fmaxf(s, 0.f);
}

__global__ void prenet2_kernel(const float* __restrict__ h1, const float* __restrict__ w2,
                               bf16_t* __restrict__ dec_bf) {
  int tid = blockIdx.x * blockDim.x + threadIdx.x;
  if (tid >= BATCH * TSTEPS * PRE) return;
  int i = tid / PRE, j = tid % PRE;
  int b = i / TSTEPS, tt = i % TSTEPS;
  const float* x = h1 + (size_t)i * PRE;
  const float* w = w2 + (size_t)j * PRE;
  float s = 0.f;
  for (int k = 0; k < PRE; ++k) s += x[k] * w[k];
  s = fmaxf(s, 0.f);
  if (tt < TSTEPS - 1) dec_bf[((size_t)(tt + 1) * BATCH + b) * PRE + j] = (bf16_t)s;
}

__global__ void pm_kernel(const float* __restrict__ memory, const float* __restrict__ wm,
                          float* __restrict__ pm) {
  int tid = blockIdx.x * blockDim.x + threadIdx.x;
  if (tid >= BATCH * LSEQ * ATTND) return;
  int i = tid / ATTND, a = tid % ATTND;
  const float* x = memory + (size_t)i * ENC;
  const float* w = wm + (size_t)a * ENC;
  float s = 0.f;
  for (int k = 0; k < ENC; ++k) s += x[k] * w[k];
  pm[tid] = s;
}

__global__ void init_state_kernel(float* qh, float* qc, float* dh, float* dc, float* ctx,
                                  float* aw, float* aws, bf16_t* qh_bf, bf16_t* dh_bf,
                                  bf16_t* ctx_bf, bf16_t* dec0) {
  int i = blockIdx.x * blockDim.x + threadIdx.x;
  if (i < BATCH * QHD) { qh[i] = 0.f; qc[i] = 0.f; dh[i] = 0.f; dc[i] = 0.f;
                         qh_bf[i] = (bf16_t)0.f; dh_bf[i] = (bf16_t)0.f; }
  if (i < BATCH * ENC) { ctx[i] = 0.f; ctx_bf[i] = (bf16_t)0.f; }
  if (i < BATCH * LSEQ) { aw[i] = 0.f; aws[i] = 0.f; }
  if (i < BATCH * PRE) { dec0[i] = (bf16_t)0.f; }
}

// ---------------- host side ----------------
extern "C" void kernel_launch(void* const* d_in, const int* in_sizes, int n_in,
                              void* d_out, int out_size, void* d_ws, size_t ws_size,
                              hipStream_t stream) {
  const float* memory    = (const float*)d_in[0];
  const float* teacher   = (const float*)d_in[2];
  const float* pre_w1    = (const float*)d_in[3];
  const float* pre_w2    = (const float*)d_in[4];
  const float* q_wih     = (const float*)d_in[5];
  const float* q_whh     = (const float*)d_in[6];
  const float* q_bih     = (const float*)d_in[7];
  const float* q_bhh     = (const float*)d_in[8];
  const float* dw_ih     = (const float*)d_in[9];
  const float* dw_hh     = (const float*)d_in[10];
  const float* db_ih     = (const float*)d_in[11];
  const float* db_hh     = (const float*)d_in[12];
  const float* attn_wq   = (const float*)d_in[13];
  const float* attn_wm   = (const float*)d_in[14];
  const float* attn_conv = (const float*)d_in[15];
  const float* attn_wloc = (const float*)d_in[16];
  const float* attn_v    = (const float*)d_in[17];
  const float* proj_w    = (const float*)d_in[18];
  const float* proj_b    = (const float*)d_in[19];
  const float* gate_w    = (const float*)d_in[20];
  const float* gate_b    = (const float*)d_in[21];

  float* out = (float*)d_out;
  float* out_mel   = out;
  float* out_align = out + (size_t)BATCH * TSTEPS * NMELS;
  float* out_stop  = out + (size_t)BATCH * TSTEPS * NMELS + (size_t)BATCH * TSTEPS * LSEQ;

  char* ws = (char*)d_ws;
  size_t off = 0;
  auto alloc = [&](size_t bytes) -> void* {
    void* p = ws + off;
    off += (bytes + 255) & ~(size_t)255;
    return p;
  };

  bf16_t* Wq_bf   = (bf16_t*)alloc((size_t)GATES * KQ * 2);
  bf16_t* Wd_bf   = (bf16_t*)alloc((size_t)GATES * KD * 2);
  bf16_t* wloc_bf = (bf16_t*)alloc((size_t)ATTND * NFILT * 2);
  bf16_t* dec_bf  = (bf16_t*)alloc((size_t)TSTEPS * BATCH * PRE * 2);
  bf16_t* qh_bf   = (bf16_t*)alloc((size_t)BATCH * QHD * 2);
  bf16_t* dh_bf   = (bf16_t*)alloc((size_t)BATCH * DHD * 2);
  bf16_t* ctx_bf  = (bf16_t*)alloc((size_t)BATCH * ENC * 2);
  bf16_t* locf_bf = (bf16_t*)alloc((size_t)BATCH * LSEQ * NFILT * 2);
  float* pm   = (float*)alloc((size_t)BATCH * LSEQ * ATTND * 4);
  float* locE = (float*)alloc((size_t)BATCH * LSEQ * ATTND * 4);
  float* h1   = (float*)alloc((size_t)BATCH * TSTEPS * PRE * 4);
  float* z    = (float*)alloc((size_t)BATCH * GATES * 4);
  float* qh   = (float*)alloc((size_t)BATCH * QHD * 4);
  float* qc   = (float*)alloc((size_t)BATCH * QHD * 4);
  float* dh   = (float*)alloc((size_t)BATCH * DHD * 4);
  float* dc   = (float*)alloc((size_t)BATCH * DHD * 4);
  float* ctx  = (float*)alloc((size_t)BATCH * ENC * 4);
  float* pq   = (float*)alloc((size_t)BATCH * ATTND * 4);
  float* aw   = (float*)alloc((size_t)BATCH * LSEQ * 4);
  float* aws  = (float*)alloc((size_t)BATCH * LSEQ * 4);
  float* energ= (float*)alloc((size_t)BATCH * LSEQ * 4);

  const int TB = 256;
  // one-time prep
  {
    int totq = GATES * KQ;
    pack_w_kernel<<<(totq + TB - 1) / TB, TB, 0, stream>>>(q_wih, PRE + ENC, q_whh, QHD, Wq_bf, totq);
    int totd = GATES * KD;
    pack_w_kernel<<<(totd + TB - 1) / TB, TB, 0, stream>>>(dw_ih, ENC + QHD, dw_hh, DHD, Wd_bf, totd);
    int nl = ATTND * NFILT;
    cvt_bf16_kernel<<<(nl + TB - 1) / TB, TB, 0, stream>>>(attn_wloc, wloc_bf, nl);
    int np = BATCH * TSTEPS * PRE;
    prenet1_kernel<<<(np + TB - 1) / TB, TB, 0, stream>>>(teacher, pre_w1, h1);
    prenet2_kernel<<<(np + TB - 1) / TB, TB, 0, stream>>>(h1, pre_w2, dec_bf);
    int npm = BATCH * LSEQ * ATTND;
    pm_kernel<<<(npm + TB - 1) / TB, TB, 0, stream>>>(memory, attn_wm, pm);
    init_state_kernel<<<(BATCH * QHD + TB - 1) / TB, TB, 0, stream>>>(
        qh, qc, dh, dc, ctx, aw, aws, qh_bf, dh_bf, ctx_bf, dec_bf);
  }

  for (int t = 0; t < TSTEPS; ++t) {
    // query LSTM gates: A = [dec_in(256) | ctx(512) | qh(1024)], W = Wq (4096 x 1792)
    wmma_gemm_rowA<<<dim3(GATES / 32, 1), 32, 0, stream>>>(
        dec_bf + (size_t)t * BATCH * PRE, PRE, ctx_bf, ENC, qh_bf, QHD, Wq_bf, KQ, z, GATES);
    lstm_point_kernel<<<(BATCH * QHD + TB - 1) / TB, TB, 0, stream>>>(z, q_bih, q_bhh, qc, qh, qh_bf);
    pq_kernel<<<(BATCH * ATTND + TB - 1) / TB, TB, 0, stream>>>(qh, attn_wq, pq);
    conv_loc_kernel<<<(BATCH * LSEQ * NFILT + TB - 1) / TB, TB, 0, stream>>>(aw, aws, attn_conv, locf_bf);
    // loc projection: (6400 x 32) @ (32 -> 128), WMMA, 400 M-tiles x 4 blocks of 2 N-tiles
    wmma_gemm_rowA<<<dim3(ATTND / 32, BATCH * LSEQ / 16), 32, 0, stream>>>(
        locf_bf, NFILT, nullptr, 0, nullptr, 0, wloc_bf, NFILT, locE, ATTND);
    energies_kernel<<<(BATCH * LSEQ + TB - 1) / TB, TB, 0, stream>>>(pq, locE, pm, attn_v, energ);
    softmax_ctx_kernel<<<BATCH, 512, 0, stream>>>(energ, memory, aw, aws, ctx, ctx_bf, out_align, t);
    // dec LSTM gates: A = [ctx(512) | qh(1024) | dh(1024)], W = Wd (4096 x 2560)
    wmma_gemm_rowA<<<dim3(GATES / 32, 1), 32, 0, stream>>>(
        ctx_bf, ENC, qh_bf, QHD, dh_bf, DHD, Wd_bf, KD, z, GATES);
    lstm_point_kernel<<<(BATCH * DHD + TB - 1) / TB, TB, 0, stream>>>(z, db_ih, db_hh, dc, dh, dh_bf);
    proj_kernel<<<(BATCH * (NMELS + 1) + TB - 1) / TB, TB, 0, stream>>>(
        dh, ctx, proj_w, proj_b, gate_w, gate_b, out_mel, out_stop, t);
  }
}
